// CausalSelfAttention_41051297415402
// MI455X (gfx1250) — compile-verified, matrix operand order checked by reading, not run on hardware
//
#include <hip/hip_runtime.h>
#include <stdint.h>

#define BB 2
#define TT 2048
#define EE 2048
#define HH 16
#define DD 128
#define BT (BB * TT)

typedef __attribute__((ext_vector_type(16))) __bf16 v16bf;
typedef __attribute__((ext_vector_type(8)))  float  v8f;
typedef unsigned long long u64;
typedef unsigned short     u16;
typedef unsigned int       u32;

union FragAB {
  v16bf bf;
  u64   q[4];
  u16   u[16];
};

__device__ __forceinline__ u16 f2bf(float f) {
  u32 u = __float_as_uint(f);
  u32 r = (u + 0x7FFFu + ((u >> 16) & 1u)) >> 16;  // RNE
  return (u16)r;
}
__device__ __forceinline__ float bf2f(u16 v) {
  return __uint_as_float((u32)v << 16);
}

__device__ __forceinline__ v8f wmma_bf16(const FragAB& a, const FragAB& b, v8f c) {
  return __builtin_amdgcn_wmma_f32_16x16x32_bf16(false, a.bf, false, b.bf,
                                                 (short)0, c, false, false);
}

// ---------------------------------------------------------------------------
// Tensor Data Mover: async 2D tile (bf16) Global -> LDS, tracked by TENSORcnt.
// D# layout per CDNA5 ISA ch.8: group0 = count/lds_addr/global_addr/type,
// group1 = data_size, tensor dims, tile dims, dim0 stride. Groups 2/3 zero
// (2D tensor). Builtin arity differs between toolchains -> clang-version gate.
// ---------------------------------------------------------------------------
#if __has_builtin(__builtin_amdgcn_tensor_load_to_lds) && \
    __has_builtin(__builtin_amdgcn_s_wait_tensorcnt)
#define HAVE_TDM 1
#else
#define HAVE_TDM 0
#endif

#if HAVE_TDM
typedef __attribute__((ext_vector_type(4))) unsigned int u32x4;
typedef __attribute__((ext_vector_type(4))) int i32x4;
typedef __attribute__((ext_vector_type(8))) int i32x8;

__device__ __forceinline__ void tdm_load_tile_bf16(
    u32 ldsOff, const void* gptr, u32 tileW /*elems*/, u32 tileH /*rows*/,
    u64 strideElems, u32 dim0 /*elems*/, u32 dim1 /*rows*/) {
  const u64 ga = (u64)gptr;
  u32x4 g0;
  g0.x = 1u;                                              // count=1, user mode
  g0.y = ldsOff;                                          // lds_addr
  g0.z = (u32)ga;                                         // global_addr[31:0]
  g0.w = (u32)((ga >> 32) & 0x01FFFFFFu) | (2u << 30);    // addr[56:32], type=2
  i32x8 g1;
  g1[0] = (int)(1u << 16);                                // data_size=1 -> 2B
  g1[1] = (int)((dim0 & 0xFFFFu) << 16);                  // tensor_dim0 lo16
  g1[2] = (int)((dim0 >> 16) | ((dim1 & 0xFFFFu) << 16)); // dim0 hi / dim1 lo
  g1[3] = (int)((dim1 >> 16) | (tileW << 16));            // dim1 hi / tile_dim0
  g1[4] = (int)(tileH & 0xFFFFu);                         // tile_dim1, dim2=0
  g1[5] = (int)(u32)strideElems;                          // dim0_stride lo32
  g1[6] = (int)((u32)(strideElems >> 32) & 0xFFFFu);      // dim0_stride hi16
  g1[7] = 0;
  const i32x4 z4 = {0, 0, 0, 0};
#if __clang_major__ >= 23
  const i32x8 z8 = {0, 0, 0, 0, 0, 0, 0, 0};
  __builtin_amdgcn_tensor_load_to_lds(g0, g1, z4, z4, z8, 0);
#else
  __builtin_amdgcn_tensor_load_to_lds(g0, g1, z4, z4, 0);
#endif
}
#define LDS_OFF(p) ((u32)(uintptr_t)(p))
#endif

// ---------------------------------------------------------------------------
// fp32 -> bf16 convert (grid-stride)
// ---------------------------------------------------------------------------
__global__ void convert_f32_to_bf16(const float* __restrict__ in,
                                    u16* __restrict__ out, int n) {
  int i = blockIdx.x * blockDim.x + threadIdx.x;
  int stride = gridDim.x * blockDim.x;
  for (; i < n; i += stride) out[i] = f2bf(in[i]);
}

// ---------------------------------------------------------------------------
// GEMM: y[m,f] = sum_e A[m,e] * W[f,e] + bias[f]
// Block = 128 threads = 4 waves; block tile = 32(M) x 64(N); each wave owns a
// 16-wide N tile and two stacked 16x16 M tiles (reuses each B fragment 2x).
// A tile (32x32 bf16, 2 KB) is TDM double-buffered in LDS; B fragments come
// straight from the row-major weight (W rows contiguous along K).
// MODE 0: bf16 -> [B,H,T,D]; MODE 1: bf16 -> [B,H,D,T]; MODE 2: fp32 -> [BT,E]
// ---------------------------------------------------------------------------
template <int MODE>
__global__ __launch_bounds__(128) void gemm_bf16_wmma(
    const u16* __restrict__ A, const u16* __restrict__ W,
    const float* __restrict__ bias, u16* __restrict__ outB,
    float* __restrict__ outF) {
  __shared__ u64 As[2][32 * 8];  // 2 x (32 rows x 32 bf16) = 4 KB

  const int lane = threadIdx.x & 31;
  const int wave = threadIdx.x >> 5;
  const int nGroups = EE / 64;  // 32
  const int mt = blockIdx.x / nGroups;
  const int nt = (blockIdx.x % nGroups) * 4 + wave;

  const int mrow0 = mt * 32;
  const int lm = lane & 15;
  const int kb = (lane >= 16) ? 8 : 0;      // K offset by lane-half (A/B layout)
  const int rowOff = (lane >= 16) ? 8 : 0;  // C/D layout row offset
  const int fcol = nt * 16 + lm;

  v8f acc0 = {}, acc1 = {};
  const int nK = EE / 32;  // 64

#if HAVE_TDM
  if (wave == 0)
    tdm_load_tile_bf16(LDS_OFF(&As[0][0]), A + (u64)mrow0 * EE,
                       32, 32, EE, EE, BT);
#endif

  for (int ki = 0; ki < nK; ++ki) {
    const int cur = ki & 1;
#if HAVE_TDM
    if (wave == 0) {
      if (ki + 1 < nK) {
        tdm_load_tile_bf16(LDS_OFF(&As[cur ^ 1][0]),
                           A + (u64)mrow0 * EE + (ki + 1) * 32,
                           32, 32, EE, EE, BT);
        __builtin_amdgcn_s_wait_tensorcnt(1);  // tile ki landed; ki+1 in flight
      } else {
        __builtin_amdgcn_s_wait_tensorcnt(0);
      }
    }
#else
    {  // cooperative fallback: 256 u64, 2 per thread
      const int r = threadIdx.x >> 2;
      const int c = (threadIdx.x & 3) * 2;
      const u64* src = (const u64*)(A + (u64)(mrow0 + r) * EE + ki * 32);
      As[cur][r * 8 + c] = src[c];
      As[cur][r * 8 + c + 1] = src[c + 1];
    }
#endif
    __syncthreads();  // A tile ki visible to all waves

    FragAB a0, a1, b;
    const u64* as = &As[cur][0];
    a0.q[0] = as[lm * 8 + (kb >> 2) + 0];
    a0.q[1] = as[lm * 8 + (kb >> 2) + 1];
    a0.q[2] = as[lm * 8 + 4 + (kb >> 2) + 0];
    a0.q[3] = as[lm * 8 + 4 + (kb >> 2) + 1];
    a1.q[0] = as[(16 + lm) * 8 + (kb >> 2) + 0];
    a1.q[1] = as[(16 + lm) * 8 + (kb >> 2) + 1];
    a1.q[2] = as[(16 + lm) * 8 + 4 + (kb >> 2) + 0];
    a1.q[3] = as[(16 + lm) * 8 + 4 + (kb >> 2) + 1];

    const u16* wrow = W + (u64)fcol * EE + ki * 32;
    b.q[0] = *(const u64*)(wrow + kb);
    b.q[1] = *(const u64*)(wrow + kb + 4);
    b.q[2] = *(const u64*)(wrow + 16 + kb);
    b.q[3] = *(const u64*)(wrow + 16 + kb + 4);

    acc0 = wmma_bf16(a0, b, acc0);
    acc1 = wmma_bf16(a1, b, acc1);
    __syncthreads();  // all reads done before buffer reuse
  }

  const float bv = bias[fcol];
#pragma unroll
  for (int half = 0; half < 2; ++half) {
    const v8f& acc = half ? acc1 : acc0;
#pragma unroll
    for (int r = 0; r < 8; ++r) {
      const int row = mrow0 + half * 16 + r + rowOff;  // bt index
      float val = acc[r] + bv;
      if (MODE == 2) {
        outF[(u64)row * EE + fcol] = val;
      } else {
        const int bidx = row / TT, t = row % TT;
        const int h = fcol >> 7, d = fcol & 127;
        if (MODE == 0)
          outB[((u64)(bidx * HH + h) * TT + t) * DD + d] = f2bf(val);
        else
          outB[((u64)(bidx * HH + h) * DD + d) * TT + t] = f2bf(val);
      }
    }
  }
}

// ---------------------------------------------------------------------------
// RoPE in-place on bf16 [B,H,T,D]; one thread per (bh, t, j<D/2) pair.
// ---------------------------------------------------------------------------
__global__ void rope_bf16(u16* __restrict__ qk, int n) {
  int i = blockIdx.x * blockDim.x + threadIdx.x;
  if (i >= n) return;
  const int j = i & 63;
  const int t = (i >> 6) % TT;
  const int bh = i / (64 * TT);
  const u64 base = (u64)bh * TT * DD + (u64)t * DD + j;
  const float x0 = bf2f(qk[base]);
  const float x1 = bf2f(qk[base + 64]);
  // ln(10000)/64 = 0.14391156831212787
  const float ang = (float)(t + 1) * __expf(-0.14391156831212787f * (float)j);
  const float c = __cosf(ang), s = __sinf(ang);
  qk[base]      = f2bf(x0 * c - x1 * s);
  qk[base + 64] = f2bf(x1 * c + x0 * s);
}

// ---------------------------------------------------------------------------
// Flash attention: block = 128 threads = 4 waves = 4 consecutive 16-row q
// tiles of one (b,h). K tile (32x128) and Vt tile (128x32) are TDM
// double-buffered in LDS and shared by all 4 waves. Online softmax in fp32;
// P bounced through per-wave LDS to reach A-fragment layout.
// ---------------------------------------------------------------------------
#define QW 4
__global__ __launch_bounds__(128) void attn_fwd(const u16* __restrict__ Q,
                                                const u16* __restrict__ K,
                                                const u16* __restrict__ Vt,
                                                u16* __restrict__ Y) {
  __shared__ u64 Kt[2][32 * 32];   // 2 x (32 keys x 128 d bf16) = 16 KB
  __shared__ u64 Vl[2][128 * 8];   // 2 x (128 d x 32 keys bf16) = 16 KB
  __shared__ u64 Pt[QW][16 * 8];   // per-wave 16x32 bf16 P tile = 4 KB

  const int qBlocks = TT / (16 * QW);  // 32
  const int qb = blockIdx.x % qBlocks;
  const int bh = blockIdx.x / qBlocks;
  const int wave = threadIdx.x >> 5;
  const int lane = threadIdx.x & 31;
  const int lm = lane & 15;
  const int kb = (lane >= 16) ? 8 : 0;
  const int rowOff = (lane >= 16) ? 8 : 0;

  const u16* Qp = Q + (u64)bh * TT * DD;
  const u16* Kp = K + (u64)bh * TT * DD;
  const u16* Vp = Vt + (u64)bh * DD * TT;
  const int q0 = (qb * QW + wave) * 16;

  // resident Q A-fragments: K=128 in 4 steps of 32
  FragAB qa[4];
#pragma unroll
  for (int ks = 0; ks < 4; ++ks) {
    const u16* qrow = Qp + (u64)(q0 + lm) * DD + ks * 32;
    qa[ks].q[0] = *(const u64*)(qrow + kb);
    qa[ks].q[1] = *(const u64*)(qrow + kb + 4);
    qa[ks].q[2] = *(const u64*)(qrow + 16 + kb);
    qa[ks].q[3] = *(const u64*)(qrow + 16 + kb + 4);
  }

  v8f o[8];
#pragma unroll
  for (int n = 0; n < 8; ++n) o[n] = {};
  float mrow[8], lrow[8];
#pragma unroll
  for (int r = 0; r < 8; ++r) { mrow[r] = -3.0e38f; lrow[r] = 0.f; }

  const float iscale = 0.08838834764831845f;  // 1/sqrt(128)
  const int nTiles = qb * 2 + 2;  // 32-key tiles covering 0..qb*64+63

#if HAVE_TDM
  if (wave == 0) {
    tdm_load_tile_bf16(LDS_OFF(&Kt[0][0]), Kp, 128, 32, DD, DD, TT);
    tdm_load_tile_bf16(LDS_OFF(&Vl[0][0]), Vp, 32, 128, TT, TT, DD);
  }
#endif

  for (int ti = 0; ti < nTiles; ++ti) {
    const int kt = ti * 32;
    const int cur = ti & 1;
#if HAVE_TDM
    if (wave == 0) {
      if (ti + 1 < nTiles) {
        tdm_load_tile_bf16(LDS_OFF(&Kt[cur ^ 1][0]),
                           Kp + (u64)(kt + 32) * DD, 128, 32, DD, DD, TT);
        tdm_load_tile_bf16(LDS_OFF(&Vl[cur ^ 1][0]),
                           Vp + (kt + 32), 32, 128, TT, TT, DD);
        __builtin_amdgcn_s_wait_tensorcnt(2);  // tile ti landed; ti+1 in flight
      } else {
        __builtin_amdgcn_s_wait_tensorcnt(0);
      }
    }
#else
    for (int i = threadIdx.x; i < 32 * 32; i += 128) {  // K tile: 1024 u64
      const int r = i >> 5, c = i & 31;
      Kt[cur][i] = *(const u64*)(Kp + (u64)(kt + r) * DD + c * 4);
    }
    for (int i = threadIdx.x; i < 128 * 8; i += 128) {  // V tile: 1024 u64
      const int r = i >> 3, c = i & 7;
      Vl[cur][i] = *(const u64*)(Vp + (u64)r * TT + kt + c * 4);
    }
#endif
    __syncthreads();  // (A) K/V tile ti visible

    v8f sL = {}, sR = {};
#pragma unroll
    for (int ks = 0; ks < 4; ++ks) {
      FragAB bL, bR;
      const u64* kt0 = &Kt[cur][0];
      bL.q[0] = kt0[lm * 32 + ks * 8 + (kb >> 2) + 0];
      bL.q[1] = kt0[lm * 32 + ks * 8 + (kb >> 2) + 1];
      bL.q[2] = kt0[lm * 32 + ks * 8 + 4 + (kb >> 2) + 0];
      bL.q[3] = kt0[lm * 32 + ks * 8 + 4 + (kb >> 2) + 1];
      bR.q[0] = kt0[(16 + lm) * 32 + ks * 8 + (kb >> 2) + 0];
      bR.q[1] = kt0[(16 + lm) * 32 + ks * 8 + (kb >> 2) + 1];
      bR.q[2] = kt0[(16 + lm) * 32 + ks * 8 + 4 + (kb >> 2) + 0];
      bR.q[3] = kt0[(16 + lm) * 32 + ks * 8 + 4 + (kb >> 2) + 1];
      sL = wmma_bf16(qa[ks], bL, sL);
      sR = wmma_bf16(qa[ks], bR, sR);
    }

    float pl[8], pr[8];
#pragma unroll
    for (int r = 0; r < 8; ++r) {
      const int row = q0 + r + rowOff;
      const int cL = kt + lm;
      float vL = (cL <= row) ? sL[r] * iscale : -3.0e38f;
      float vR = (cL + 16 <= row) ? sR[r] * iscale : -3.0e38f;
      float v = fmaxf(vL, vR);
      v = fmaxf(v, __shfl_xor(v, 1, 32));
      v = fmaxf(v, __shfl_xor(v, 2, 32));
      v = fmaxf(v, __shfl_xor(v, 4, 32));
      v = fmaxf(v, __shfl_xor(v, 8, 32));
      const float mnew = fmaxf(mrow[r], v);
      const float alpha = __expf(mrow[r] - mnew);
      const float eL = __expf(vL - mnew);
      const float eR = __expf(vR - mnew);
      float rs = eL + eR;
      rs += __shfl_xor(rs, 1, 32);
      rs += __shfl_xor(rs, 2, 32);
      rs += __shfl_xor(rs, 4, 32);
      rs += __shfl_xor(rs, 8, 32);
      lrow[r] = lrow[r] * alpha + rs;
      mrow[r] = mnew;
      pl[r] = eL;
      pr[r] = eR;
#pragma unroll
      for (int n = 0; n < 8; ++n) o[n][r] *= alpha;
    }

    {  // C-layout -> A-fragment layout via per-wave LDS bounce
      u16* pt = (u16*)&Pt[wave][0];
#pragma unroll
      for (int r = 0; r < 8; ++r) {
        const int row = r + rowOff;
        pt[row * 32 + lm] = f2bf(pl[r]);
        pt[row * 32 + 16 + lm] = f2bf(pr[r]);
      }
    }
    __syncthreads();  // (B) P visible
    FragAB pa;
    pa.q[0] = Pt[wave][lm * 8 + (kb >> 2) + 0];
    pa.q[1] = Pt[wave][lm * 8 + (kb >> 2) + 1];
    pa.q[2] = Pt[wave][lm * 8 + 4 + (kb >> 2) + 0];
    pa.q[3] = Pt[wave][lm * 8 + 4 + (kb >> 2) + 1];

#pragma unroll
    for (int n = 0; n < 8; ++n) {
      FragAB vb;
      const u64* vt0 = &Vl[cur][0];
      vb.q[0] = vt0[(n * 16 + lm) * 8 + (kb >> 2) + 0];
      vb.q[1] = vt0[(n * 16 + lm) * 8 + (kb >> 2) + 1];
      vb.q[2] = vt0[(n * 16 + lm) * 8 + 4 + (kb >> 2) + 0];
      vb.q[3] = vt0[(n * 16 + lm) * 8 + 4 + (kb >> 2) + 1];
      o[n] = wmma_bf16(pa, vb, o[n]);
    }
    __syncthreads();  // (C) buffers free for next TDM / P rewrite
  }

  // epilogue: Y[b*T + t, h*128 + d] bf16
  const int b = bh / HH, h = bh % HH;
#pragma unroll
  for (int r = 0; r < 8; ++r) {
    const int t = q0 + r + rowOff;
    const u64 bt = (u64)b * TT + t;
    const float inv = 1.0f / lrow[r];
#pragma unroll
    for (int n = 0; n < 8; ++n) {
      const int d = n * 16 + lm;
      Y[bt * EE + (u64)h * 128 + d] = f2bf(o[n][r] * inv);
    }
  }
}

// ---------------------------------------------------------------------------
extern "C" void kernel_launch(void* const* d_in, const int* in_sizes, int n_in,
                              void* d_out, int out_size, void* d_ws,
                              size_t ws_size, hipStream_t stream) {
  const float* x  = (const float*)d_in[0];
  const float* Wq = (const float*)d_in[1];
  const float* bq = (const float*)d_in[2];
  const float* Wk = (const float*)d_in[3];
  const float* bk = (const float*)d_in[4];
  const float* Wv = (const float*)d_in[5];
  const float* bv = (const float*)d_in[6];
  const float* Wo = (const float*)d_in[7];
  const float* bo = (const float*)d_in[8];
  float* out = (float*)d_out;

  char* ws = (char*)d_ws;
  size_t off = 0;
  auto alloc = [&](size_t bytes) {
    char* p = ws + off;
    off += (bytes + 255) & ~(size_t)255;
    return p;
  };
  u16* xb  = (u16*)alloc((size_t)BT * EE * 2);
  u16* wqb = (u16*)alloc((size_t)EE * EE * 2);
  u16* wkb = (u16*)alloc((size_t)EE * EE * 2);
  u16* wvb = (u16*)alloc((size_t)EE * EE * 2);
  u16* wob = (u16*)alloc((size_t)EE * EE * 2);
  u16* Qb  = (u16*)alloc((size_t)BB * HH * TT * DD * 2);
  u16* Kb  = (u16*)alloc((size_t)BB * HH * TT * DD * 2);
  u16* Vt  = (u16*)alloc((size_t)BB * HH * DD * TT * 2);
  u16* Yb  = (u16*)alloc((size_t)BT * EE * 2);
  (void)ws_size;

  // 1) fp32 -> bf16
  convert_f32_to_bf16<<<2048, 256, 0, stream>>>(x, xb, BT * EE);
  convert_f32_to_bf16<<<2048, 256, 0, stream>>>(Wq, wqb, EE * EE);
  convert_f32_to_bf16<<<2048, 256, 0, stream>>>(Wk, wkb, EE * EE);
  convert_f32_to_bf16<<<2048, 256, 0, stream>>>(Wv, wvb, EE * EE);
  convert_f32_to_bf16<<<2048, 256, 0, stream>>>(Wo, wob, EE * EE);

  // 2) projections (32x64 block tiles)
  dim3 gGemm((BT / 32) * (EE / 64));
  gemm_bf16_wmma<0><<<gGemm, 128, 0, stream>>>(xb, wqb, bq, Qb, nullptr);
  gemm_bf16_wmma<0><<<gGemm, 128, 0, stream>>>(xb, wkb, bk, Kb, nullptr);
  gemm_bf16_wmma<1><<<gGemm, 128, 0, stream>>>(xb, wvb, bv, Vt, nullptr);

  // 3) RoPE on Q and K
  const int nR = BB * HH * TT * (DD / 2);
  rope_bf16<<<(nR + 255) / 256, 256, 0, stream>>>(Qb, nR);
  rope_bf16<<<(nR + 255) / 256, 256, 0, stream>>>(Kb, nR);

  // 4) flash attention (4 q-tiles per block share K/V tiles)
  attn_fwd<<<BB * HH * (TT / 64), 128, 0, stream>>>(Qb, Kb, Vt, Yb);

  // 5) output projection -> fp32
  gemm_bf16_wmma<2><<<gGemm, 128, 0, stream>>>(Yb, wob, bo, nullptr, out);
}
